// FraudEdgeGAT_79551384257040
// MI455X (gfx1250) — matrix operand order repaired and checked by reading, not compiled, not run on hardware
//
#include <hip/hip_runtime.h>
#include <hip/hip_bf16.h>

typedef __attribute__((ext_vector_type(16))) _Float16 v16h;
typedef __attribute__((ext_vector_type(8)))  _Float16 h8;
typedef __attribute__((ext_vector_type(4)))  _Float16 h4;
typedef __attribute__((ext_vector_type(4)))  float    f4;
typedef __attribute__((ext_vector_type(8)))  float    v8f;

#define N_NODES 100000
#define N_EDGES 500000
#define F_NODE  128
#define F_EDGE  32
#define HEADS   4
#define CDIM    64
#define HC      256
#define OUTF    32
#define NEG_SLOPE 0.2f
// 1/sqrt(1 + 1e-5)
#define BN_SCALE 0.9999950000374994f

// ---------------------------------------------------------------------------
// Swizzle an fp32 [K, srcNout] weight matrix into the per-lane WMMA B-fragment
// layout as f16:  Bsw[(kkc*dstNumNT + ntBase+ntL)*512 + lane*16 + i]
// where for lane: n = ntL*16 + (lane&15), k = kkc*32 + 16*(lane>>4) + i.
// A GEMM B-fragment then becomes two contiguous 16B loads, no converts.
// ---------------------------------------------------------------------------
__global__ __launch_bounds__(256)
void swizzleB_kernel(const float* __restrict__ B, _Float16* __restrict__ Bsw,
                     int K, int srcNout, int dstNumNT, int ntBase) {
  int idx = blockIdx.x * blockDim.x + threadIdx.x;
  int srcNT = srcNout >> 4;
  int totalHalves = (K >> 5) * srcNT * 512;
  if (idx >= totalHalves) return;
  int i    = idx & 15;
  int lane = (idx >> 4) & 31;
  int t    = idx >> 9;            // kkc * srcNT + ntL
  int kkc  = t / srcNT;
  int ntL  = t % srcNT;
  int n = ntL * 16 + (lane & 15);
  int k = kkc * 32 + ((lane >> 4) << 4) + i;
  size_t dst = ((size_t)(kkc * dstNumNT + ntBase + ntL) << 9) + (lane << 4) + i;
  Bsw[dst] = (_Float16)B[(size_t)k * srcNout + n];
}

// ---------------------------------------------------------------------------
// WMMA GEMM: C[M,NOUT] = epi(A[M,K] @ B[K,NOUT] + bias)
// fp32 A/C, pre-swizzled f16 B. Block = 256 threads (8 waves); STRIPS 16-row
// A strips staged in LDS as f16 (vectorized b128 load + cvt_pk + b64 store).
// Hot loop per 16x16x32 tile: 2x ds_load_b128 + 2x global_load(16B) + 1 WMMA.
// ---------------------------------------------------------------------------
template <int K, int NOUT, int STRIPS, int EPI>
__global__ __launch_bounds__(256)
void gemm_wmma_f16(const float* __restrict__ A, const _Float16* __restrict__ Bsw,
                   const float* __restrict__ bias, float* __restrict__ C,
                   int M) {
  static_assert((K % 32) == 0 && K <= 256, "K");
  static_assert((NOUT % 16) == 0, "NOUT");
  constexpr int numNT = NOUT / 16;
  constexpr int SLOTS = STRIPS * numNT;
  static_assert(SLOTS >= 8, "fill all 8 waves");

  __shared__ _Float16 smA[STRIPS * 16 * K];

  const int row0 = blockIdx.x * (STRIPS * 16);

  // Stage (STRIPS*16) x K of A into LDS as f16, 4 elements per thread-step.
  constexpr int TOT4 = (STRIPS * 16 * K) / 4;
#pragma unroll 2
  for (int q = threadIdx.x; q < TOT4; q += 256) {
    int idx = q * 4;
    int r = idx / K;                   // K power of two -> shift
    int gr = row0 + r;
    f4 v = {0.f, 0.f, 0.f, 0.f};
    if (gr < M) v = *(const f4*)&A[(size_t)gr * K + (idx % K)];
    h4 hv = {(_Float16)v.x, (_Float16)v.y, (_Float16)v.z, (_Float16)v.w};
    *(h4*)&smA[idx] = hv;
  }
  __syncthreads();

  const int lane = threadIdx.x & 31;
  const int w    = threadIdx.x >> 5;
  const int koff = (lane >> 4) << 4;    // A K offset (0 or 16)

  for (int slot = w; slot < SLOTS; slot += 8) {
    const int strip = slot / numNT;
    const int nt    = slot % numNT;
    const int mloc  = strip * 16 + (lane & 15);
    const int n0    = nt * 16 + (lane & 15);

    v8f acc = {0.f, 0.f, 0.f, 0.f, 0.f, 0.f, 0.f, 0.f};

#pragma unroll
    for (int kk = 0; kk < K; kk += 32) {
      // A fragment (16x32 f16), contiguous:
      // halves 0..15 -> K = kk+koff+0..15
      const _Float16* ap = &smA[mloc * K + kk + koff];
      h8 alo = *(const h8*)ap;
      h8 ahi = *(const h8*)(ap + 8);
      v16h af;
#pragma unroll
      for (int i = 0; i < 8; ++i) { af[i] = alo[i]; af[i + 8] = ahi[i]; }

      // B fragment: pre-swizzled, contiguous 16 halves per lane.
      const _Float16* bp =
          &Bsw[(((size_t)(kk >> 5) * numNT + nt) << 9) + (lane << 4)];
      h8 blo = *(const h8*)bp;
      h8 bhi = *(const h8*)(bp + 8);
      v16h bf;
#pragma unroll
      for (int i = 0; i < 8; ++i) { bf[i] = blo[i]; bf[i + 8] = bhi[i]; }

      acc = __builtin_amdgcn_wmma_f32_16x16x32_f16(
          false, af, false, bf, (short)0, acc, false, false);
    }

    // D layout: lane -> N = lane&15, M = strip*16 + 8*(lane>>4) + v
    const float bv = (EPI != 0) ? bias[n0] : 0.0f;
    const int mb    = row0 + strip * 16;
    const int mbase = mb + ((lane >> 4) << 3);
    if (mb + 16 <= M) {
#pragma unroll
      for (int v = 0; v < 8; ++v) {
        float t = acc[v] + bv;
        if (EPI == 1) t = fmaxf(t, 0.0f);
        C[(size_t)(mbase + v) * NOUT + n0] = t;
      }
    } else {
#pragma unroll
      for (int v = 0; v < 8; ++v) {
        if (mbase + v < M) {
          float t = acc[v] + bv;
          if (EPI == 1) t = fmaxf(t, 0.0f);
          C[(size_t)(mbase + v) * NOUT + n0] = t;
        }
      }
    }
  }
}

// ---------------------------------------------------------------------------
// Per-(node,head) attention coefficients: a_s/a_d = sum_c xh[n,h,c]*att[h,c]
// ---------------------------------------------------------------------------
__global__ __launch_bounds__(256)
void asd_kernel(const float* __restrict__ xh,
                const float* __restrict__ att_s, const float* __restrict__ att_d,
                float* __restrict__ a_s, float* __restrict__ a_d, int NH) {
  int wid = (int)((blockIdx.x * blockDim.x + threadIdx.x) >> 5);
  if (wid >= NH) return;
  int lane = threadIdx.x & 31;
  int node = wid >> 2;
  int hd   = wid & 3;
  const float* row = xh + (size_t)node * HC + hd * CDIM;
  const float* vs  = att_s + hd * CDIM;
  const float* vd  = att_d + hd * CDIM;
  float x0 = row[lane], x1 = row[lane + 32];
  float ss = x0 * vs[lane] + x1 * vs[lane + 32];
  float sd = x0 * vd[lane] + x1 * vd[lane + 32];
#pragma unroll
  for (int off = 16; off; off >>= 1) {
    ss += __shfl_xor(ss, off, 32);
    sd += __shfl_xor(sd, off, 32);
  }
  if (lane == 0) { a_s[wid] = ss; a_d[wid] = sd; }
}

// Monotone float<->uint mapping so unsigned atomicMax implements float max.
__device__ __forceinline__ unsigned f2mono(float f) {
  unsigned u = __float_as_uint(f);
  return (u & 0x80000000u) ? ~u : (u | 0x80000000u);
}
__device__ __forceinline__ float mono2f(unsigned m) {
  unsigned u = (m & 0x80000000u) ? (m & 0x7FFFFFFFu) : ~m;
  return __uint_as_float(u);
}

// e = leaky_relu(a_s[src] + a_d[dst]); segment-max into m[dst] (mono atomicMax)
__global__ __launch_bounds__(256)
void edge_max_kernel(const int* __restrict__ src, const int* __restrict__ dst,
                     const float* __restrict__ a_s, const float* __restrict__ a_d,
                     float* __restrict__ e_raw, unsigned* __restrict__ m_mono,
                     int EH) {
  int idx = blockIdx.x * blockDim.x + threadIdx.x;
  if (idx >= EH) return;
  int e = idx >> 2, hd = idx & 3;
  float v = a_s[src[e] * HEADS + hd] + a_d[dst[e] * HEADS + hd];
  v = (v > 0.0f) ? v : NEG_SLOPE * v;
  e_raw[idx] = v;
  atomicMax(&m_mono[dst[e] * HEADS + hd], f2mono(v));
}

// ex = exp(e - m[dst]); den[dst] += ex
__global__ __launch_bounds__(256)
void edge_exp_kernel(const int* __restrict__ dst,
                     float* __restrict__ e_raw, const unsigned* __restrict__ m_mono,
                     float* __restrict__ den, int EH) {
  int idx = blockIdx.x * blockDim.x + threadIdx.x;
  if (idx >= EH) return;
  int e = idx >> 2, hd = idx & 3;
  int d = dst[e] * HEADS + hd;
  float x = __expf(e_raw[idx] - mono2f(m_mono[d]));
  e_raw[idx] = x;
  atomicAdd(&den[d], x);
}

// agg[dst] += xh[src] * alpha   (wave per edge, 256 channels / 8 steps)
__global__ __launch_bounds__(256)
void agg_kernel(const int* __restrict__ src, const int* __restrict__ dst,
                const float* __restrict__ xh, const float* __restrict__ ex,
                const float* __restrict__ den, float* __restrict__ agg, int E) {
  int wid = (int)((blockIdx.x * blockDim.x + threadIdx.x) >> 5);
  if (wid >= E) return;
  int lane = threadIdx.x & 31;
  int s = src[wid], d = dst[wid];
  float alpha[HEADS];
#pragma unroll
  for (int hd = 0; hd < HEADS; ++hd)
    alpha[hd] = ex[(size_t)wid * HEADS + hd] / (den[(size_t)d * HEADS + hd] + 1e-16f);
  const float* xrow = xh + (size_t)s * HC;
  float* arow = agg + (size_t)d * HC;
#pragma unroll
  for (int st = 0; st < 8; ++st) {
    int c = st * 32 + lane;                 // head = c>>6 = st>>1 (uniform/step)
    atomicAdd(&arow[c], xrow[c] * alpha[st >> 1]);
  }
}

// hn = relu((agg + bg) * g*BN_SCALE + be);  h = residual ? h + hn : hn
template <bool RES>
__global__ __launch_bounds__(256)
void bn_kernel(const float* __restrict__ agg, const float* __restrict__ bg,
               const float* __restrict__ g, const float* __restrict__ be,
               const float* __restrict__ hin, float* __restrict__ hout,
               size_t total) {
  size_t i = (size_t)blockIdx.x * blockDim.x + threadIdx.x;
  if (i >= total) return;
  int c = (int)(i & (HC - 1));
  float t = agg[i] + bg[c];
  t = t * (g[c] * BN_SCALE) + be[c];
  t = fmaxf(t, 0.0f);
  hout[i] = RES ? (hin[i] + t) : t;
}

// out[e] = b2 + sum_j relu(zsd[src,j] + zsd[dst,32+j] + ze[e,j] + b1[j]) * W2[j]
__global__ __launch_bounds__(256)
void final_edge_kernel(const int* __restrict__ src, const int* __restrict__ dst,
                       const float* __restrict__ zsd, const float* __restrict__ ze,
                       const float* __restrict__ b1, const float* __restrict__ W2,
                       const float* __restrict__ b2, float* __restrict__ out, int E) {
  int wid = (int)((blockIdx.x * blockDim.x + threadIdx.x) >> 5);
  if (wid >= E) return;
  int lane = threadIdx.x & 31;   // lane = output feature j (OUTF==32)
  int s = src[wid], d = dst[wid];
  float v = zsd[(size_t)s * 64 + lane] + zsd[(size_t)d * 64 + 32 + lane] +
            ze[(size_t)wid * OUTF + lane] + b1[lane];
  v = fmaxf(v, 0.0f);
  float p = v * W2[lane];
#pragma unroll
  for (int off = 16; off; off >>= 1) p += __shfl_xor(p, off, 32);
  if (lane == 0) out[wid] = p + b2[0];
}

// ---------------------------------------------------------------------------
extern "C" void kernel_launch(void* const* d_in, const int* in_sizes, int n_in,
                              void* d_out, int out_size, void* d_ws, size_t ws_size,
                              hipStream_t stream) {
  const int N = N_NODES, E = N_EDGES;

  const float* x    = (const float*)d_in[0];
  const int*   eidx = (const int*)d_in[1];
  const float* eatt = (const float*)d_in[2];
  const float* Wp   = (const float*)d_in[3];
  const float* bp   = (const float*)d_in[4];
  const float* Wg[3]  = {(const float*)d_in[5],  (const float*)d_in[11], (const float*)d_in[17]};
  const float* as_[3] = {(const float*)d_in[6],  (const float*)d_in[12], (const float*)d_in[18]};
  const float* ad_[3] = {(const float*)d_in[7],  (const float*)d_in[13], (const float*)d_in[19]};
  const float* bg_[3] = {(const float*)d_in[8],  (const float*)d_in[14], (const float*)d_in[20]};
  const float* g_[3]  = {(const float*)d_in[9],  (const float*)d_in[15], (const float*)d_in[21]};
  const float* be_[3] = {(const float*)d_in[10], (const float*)d_in[16], (const float*)d_in[22]};
  const float* W1 = (const float*)d_in[23];
  const float* b1 = (const float*)d_in[24];
  const float* W2 = (const float*)d_in[25];
  const float* b2 = (const float*)d_in[26];

  const int* src = eidx;
  const int* dst = eidx + E;

  // Workspace layout (floats)
  float* ws = (float*)d_ws;
  size_t off = 0;
  float* h0  = ws + off; off += (size_t)N * CDIM;   // projected features [N,64]
  float* h   = ws + off; off += (size_t)N * HC;     // running features   [N,256]
  float* xh  = ws + off; off += (size_t)N * HC;     // per-layer h @ Wg   [N,256]
  float* agg = ws + off; off += (size_t)N * HC;     // aggregation / reused as ze
  float* zsd = ws + off; off += (size_t)N * 64;     // fused [zs | zd]
  float* a_s = ws + off; off += (size_t)N * HEADS;
  float* a_d = ws + off; off += (size_t)N * HEADS;
  unsigned* m_mono = (unsigned*)(ws + off); off += (size_t)N * HEADS;
  float* den = ws + off; off += (size_t)N * HEADS;
  float* ex  = ws + off; off += (size_t)E * HEADS;  // raw e then softmax numerators

  // f16 swizzled-weight region (tail of workspace)
  _Float16* wsw = (_Float16*)(ws + off);
  size_t hoff = 0;
  _Float16* BswP = wsw + hoff; hoff += (size_t)F_NODE * CDIM;  // 128x64
  _Float16* Bsw0 = wsw + hoff; hoff += (size_t)CDIM * HC;      // 64x256
  _Float16* Bsw1 = wsw + hoff; hoff += (size_t)HC * HC;        // 256x256
  _Float16* Bsw2 = wsw + hoff; hoff += (size_t)HC * HC;        // 256x256
  _Float16* BswZ = wsw + hoff; hoff += (size_t)HC * 64;        // 256x64 (zs|zd)
  _Float16* BswE = wsw + hoff; hoff += (size_t)F_EDGE * OUTF;  // 32x32

  const int TB = 256;
  auto swz = [&](const float* B, _Float16* Bs, int K, int srcNout, int dstNT, int ntBase) {
    int total = (K >> 5) * (srcNout >> 4) * 512;
    swizzleB_kernel<<<(total + TB - 1) / TB, TB, 0, stream>>>(B, Bs, K, srcNout, dstNT, ntBase);
  };
  swz(Wp,    BswP, F_NODE, CDIM, CDIM / 16, 0);
  swz(Wg[0], Bsw0, CDIM,   HC,   HC / 16,   0);
  swz(Wg[1], Bsw1, HC,     HC,   HC / 16,   0);
  swz(Wg[2], Bsw2, HC,     HC,   HC / 16,   0);
  swz(W1,                    BswZ, HC, OUTF, 4, 0);  // zs -> tiles 0..1
  swz(W1 + (size_t)HC * OUTF, BswZ, HC, OUTF, 4, 2); // zd -> tiles 2..3
  swz(W1 + (size_t)2 * HC * OUTF, BswE, F_EDGE, OUTF, 2, 0);

  // h0 = relu(x @ Wp + bp)
  gemm_wmma_f16<F_NODE, CDIM, 2, 1><<<(N + 31) / 32, TB, 0, stream>>>(x, BswP, bp, h0, N);

  for (int l = 0; l < 3; ++l) {
    if (l == 0)
      gemm_wmma_f16<CDIM, HC, 1, 0><<<(N + 15) / 16, TB, 0, stream>>>(h0, Bsw0, bp, xh, N);
    else
      gemm_wmma_f16<HC, HC, 1, 0><<<(N + 15) / 16, TB, 0, stream>>>(h, l == 1 ? Bsw1 : Bsw2, bp, xh, N);

    hipMemsetAsync(m_mono, 0, (size_t)N * HEADS * sizeof(unsigned), stream);
    hipMemsetAsync(den, 0, (size_t)N * HEADS * sizeof(float), stream);
    hipMemsetAsync(agg, 0, (size_t)N * HC * sizeof(float), stream);

    int nWaves = N * HEADS;
    asd_kernel<<<(nWaves * 32 + TB - 1) / TB, TB, 0, stream>>>(xh, as_[l], ad_[l], a_s, a_d, nWaves);

    int EH = E * HEADS;
    edge_max_kernel<<<(EH + TB - 1) / TB, TB, 0, stream>>>(src, dst, a_s, a_d, ex, m_mono, EH);
    edge_exp_kernel<<<(EH + TB - 1) / TB, TB, 0, stream>>>(dst, ex, m_mono, den, EH);

    agg_kernel<<<((size_t)E * 32 + TB - 1) / TB, TB, 0, stream>>>(src, dst, xh, ex, den, agg, E);

    size_t total = (size_t)N * HC;
    if (l == 0)
      bn_kernel<false><<<(total + TB - 1) / TB, TB, 0, stream>>>(agg, bg_[l], g_[l], be_[l], h, h, total);
    else
      bn_kernel<true><<<(total + TB - 1) / TB, TB, 0, stream>>>(agg, bg_[l], g_[l], be_[l], h, h, total);
  }

  // zsd = h @ [W1s | W1d]  (one pass over h);  ze = edge_attr @ W1e
  float* ze = agg;  // reuse: E*32 floats fit in N*256 region
  gemm_wmma_f16<HC, 64, 2, 0><<<(N + 31) / 32, TB, 0, stream>>>(h, BswZ, bp, zsd, N);
  gemm_wmma_f16<F_EDGE, OUTF, 4, 0><<<(E + 63) / 64, TB, 0, stream>>>(eatt, BswE, bp, ze, E);

  // out[e] = relu(zsd[src,:32]+zsd[dst,32:]+ze+b1) @ W2 + b2
  final_edge_kernel<<<((size_t)E * 32 + TB - 1) / TB, TB, 0, stream>>>(
      src, dst, zsd, ze, b1, W2, b2, (float*)d_out, E);
}
